// ORAttention_30090540876305
// MI455X (gfx1250) — matrix operand order repaired and checked by reading, not compiled, not run on hardware
//
#include <hip/hip_runtime.h>
#include <hip/hip_bf16.h>

typedef _Float16 v16h __attribute__((ext_vector_type(16)));
typedef _Float16 v8h  __attribute__((ext_vector_type(8)));
typedef float    v8f  __attribute__((ext_vector_type(8)));

#define DIMC   512
#define HEADS  8
#define DHEAD  64
#define INNERC 512
#define SCALEF 0.125f   // 64^-0.5

// ---------------------------------------------------------------------------
// WMMA fragment loaders (wave32; A and B share contiguous k = 16g+s order, not the ISA table)
// A (16x32 f16): lane m = lane%16, g = lane/16; halves 0..15 -> K = k0+16g..
//                (two contiguous v8h loads)
// B (32x16 f16): lane n = lane%16, halves h -> K = k0+16g+h (one v16h load)
//                operand must be stored transposed: T[n][k], ld = K stride
// C/D (16x16 f32): lane n = lane%16, VGPR r -> row m = r + 8*(lane/16)
// ---------------------------------------------------------------------------
__device__ __forceinline__ v16h load_afrag(const _Float16* base, int ld, int m0, int k0) {
  int lane = threadIdx.x & 31;
  int m = lane & 15, g = lane >> 4;
  const _Float16* p = base + (size_t)(m0 + m) * ld + k0;
  v8h lo = *(const v8h*)(p + 16 * g);
  v8h hi = *(const v8h*)(p + 16 * g + 8);
  v16h a;
#pragma unroll
  for (int i = 0; i < 8; ++i) { a[i] = lo[i]; a[8 + i] = hi[i]; }
  return a;
}

__device__ __forceinline__ v16h load_bfrag(const _Float16* base, int ld, int n0, int k0) {
  int lane = threadIdx.x & 31;
  int n = lane & 15, g = lane >> 4;
  return *(const v16h*)(base + (size_t)(n0 + n) * ld + k0 + 16 * g);
}

// B fragment straight out of an LDS panel laid out [64 cols][32 halves]
__device__ __forceinline__ v16h load_bfrag_lds(const _Float16* bp, int n0) {
  int lane = threadIdx.x & 31;
  int n = lane & 15, g = lane >> 4;
  return *(const v16h*)(bp + (n0 + n) * 32 + 16 * g);
}

__device__ __forceinline__ v8f wmma_f16(v16h a, v16h b, v8f c) {
  return __builtin_amdgcn_wmma_f32_16x16x32_f16(false, a, false, b, (short)0, c,
                                                false, false);
}

// ---------------------------------------------------------------------------
// CDNA5 async-to-LDS staging (GLOBAL_LOAD_ASYNC_TO_LDS_B128, ASYNCcnt).
// VDST VGPR = LDS byte offset, VADDR = 64-bit global address (GV mode).
// ---------------------------------------------------------------------------
__device__ __forceinline__ void async_copy_b128(const _Float16* gsrc, _Float16* ldst) {
  unsigned loff = (unsigned)(uintptr_t)ldst;  // low 32 bits = LDS offset
  asm volatile("global_load_async_to_lds_b128 %0, %1, off"
               :: "v"(loff), "v"(gsrc) : "memory");
}

__device__ __forceinline__ void wait_async0() {
  asm volatile("s_wait_asynccnt 0x0" ::: "memory");
}

// Stage a 64x32-half B panel (cols n_base..+63, k chunk k0..+31) into LDS.
// Straight-line: thread owns row tid/2, half-range (tid&1)*16..+15,
// issuing exactly two b128 async copies (no divergent loop in the hot path).
__device__ __forceinline__ void stage_bpanel(const _Float16* __restrict__ Bt, int K,
                                             int n_base, int k0, _Float16* buf, int tid) {
  int n = tid >> 1;               // 64 panel rows, 2 threads per row
  int kk = (tid & 1) * 16;        // 16-half slice per thread
  const _Float16* gsrc = Bt + (size_t)(n_base + n) * K + k0 + kk;
  _Float16* ldst = buf + n * 32 + kk;
  async_copy_b128(gsrc, ldst);
  async_copy_b128(gsrc + 8, ldst + 8);
}

// ---------------------------------------------------------------------------
// Fused LayerNorm + f16 cast for x and ORquery. One wave32 per 512-wide row.
// ---------------------------------------------------------------------------
__global__ void ln_f16_kernel(const float* __restrict__ x, const float* __restrict__ orq,
                              const float* __restrict__ w, const float* __restrict__ bia,
                              _Float16* __restrict__ xn, _Float16* __restrict__ orqn,
                              int rows) {
  int gwave = (int)((blockIdx.x * blockDim.x + threadIdx.x) >> 5);
  int lane = threadIdx.x & 31;
  if (gwave >= 2 * rows) return;
  const float* src;
  _Float16* dst;
  if (gwave < rows) { src = x   + (size_t)gwave * DIMC;          dst = xn   + (size_t)gwave * DIMC; }
  else              { src = orq + (size_t)(gwave - rows) * DIMC; dst = orqn + (size_t)(gwave - rows) * DIMC; }

  float v[16];
  float s = 0.f;
#pragma unroll
  for (int i = 0; i < 16; ++i) { v[i] = src[lane + 32 * i]; s += v[i]; }
#pragma unroll
  for (int off = 16; off; off >>= 1) s += __shfl_xor(s, off);
  float mu = s * (1.f / DIMC);
  float var = 0.f;
#pragma unroll
  for (int i = 0; i < 16; ++i) { float d = v[i] - mu; var += d * d; }
#pragma unroll
  for (int off = 16; off; off >>= 1) var += __shfl_xor(var, off);
  float inv = rsqrtf(var * (1.f / DIMC) + 1e-5f);
#pragma unroll
  for (int i = 0; i < 16; ++i) {
    int c = lane + 32 * i;
    dst[c] = (_Float16)((v[i] - mu) * inv * w[c] + bia[c]);
  }
}

// ---------------------------------------------------------------------------
// Weight transpose + f16 cast: w[K][Ncol] -> wt[Ncol][K]
// ---------------------------------------------------------------------------
__global__ void wtrans_kernel(const float* __restrict__ w, _Float16* __restrict__ wt,
                              int K, int Ncol) {
  size_t idx = (size_t)blockIdx.x * blockDim.x + threadIdx.x;
  if (idx >= (size_t)K * Ncol) return;
  int n = (int)(idx / K);
  int k = (int)(idx % K);
  wt[idx] = (_Float16)w[(size_t)k * Ncol + n];
}

// ---------------------------------------------------------------------------
// WMMA GEMM: C[M][Ncol] = A[M][K] * Bt^T, Bt stored [Ncol][K] (transposed).
// Block = 128 threads = 4 waves; block tile 128 rows x 64 cols; each wave owns
// a 32x64 strip (2 A frags, 8 accumulators, 8 WMMA per K chunk).
// The shared 64x32 B panel is double-buffered in LDS via async-to-LDS copies.
// mode 0: scatter qkv -> Q[B,H,N,64] (o0), K[B,H,N,64] (o1), V^T[B,H,64,N] (o2)
// mode 1: head-split f16 -> o0[B,H,N,64]
// mode 2: f32 row-major -> oF[M][Ncol]
// ---------------------------------------------------------------------------
__global__ void wmma_gemm_kernel(const _Float16* __restrict__ A, const _Float16* __restrict__ Bt,
                                 int M, int Ncol, int K, int nseq, int mode,
                                 _Float16* __restrict__ o0, _Float16* __restrict__ o1,
                                 _Float16* __restrict__ o2, float* __restrict__ oF) {
  __shared__ __attribute__((aligned(32))) _Float16 bpanel[2][64 * 32];
  int tid = threadIdx.x;
  int warp = tid >> 5;
  int lane = tid & 31;
  int nl = lane & 15, g = lane >> 4;
  int n_base = blockIdx.x * 64;
  int m0 = blockIdx.y * 128 + warp * 32;

  v8f acc[8] = {};
  int nk = K / 32;
  stage_bpanel(Bt, K, n_base, 0, bpanel[0], tid);

  for (int kc = 0; kc < nk; ++kc) {
    int k0 = kc * 32;
    wait_async0();
    __syncthreads();
    if (kc + 1 < nk) stage_bpanel(Bt, K, n_base, k0 + 32, bpanel[(kc + 1) & 1], tid);

    v16h a0 = load_afrag(A, K, m0, k0);
    v16h a1 = load_afrag(A, K, m0 + 16, k0);
    const _Float16* bp = bpanel[kc & 1];
#pragma unroll
    for (int t = 0; t < 4; ++t) {
      v16h b = load_bfrag_lds(bp, 16 * t);
      acc[t]     = wmma_f16(a0, b, acc[t]);
      acc[4 + t] = wmma_f16(a1, b, acc[4 + t]);
    }
  }

#pragma unroll
  for (int p2 = 0; p2 < 2; ++p2) {
#pragma unroll
    for (int t = 0; t < 4; ++t) {
      int c = n_base + 16 * t + nl;
#pragma unroll
      for (int r = 0; r < 8; ++r) {
        int row = m0 + 16 * p2 + r + 8 * g;
        float fval = acc[4 * p2 + t][r];
        if (mode == 2) {
          oF[(size_t)row * Ncol + c] = fval;
        } else {
          int sec = (mode == 0) ? (c >> 9) : 0;  // /512 : which of Q,K,V
          int cc = c & 511;
          int hh = cc >> 6, d = cc & 63;
          int bidx = row / nseq, i = row % nseq;
          size_t bhh = (size_t)(bidx * HEADS + hh);
          _Float16 hval = (_Float16)fval;
          if (sec == 0)      o0[(bhh * nseq + i) * DHEAD + d] = hval;   // Q natural
          else if (sec == 1) o1[(bhh * nseq + i) * DHEAD + d] = hval;   // K natural
          else               o2[(bhh * DHEAD + d) * nseq + i] = hval;   // V transposed
        }
      }
    }
  }
}

// ---------------------------------------------------------------------------
// Flash-attention: block = 4 waves, each wave owns a 16-query tile, streams
// keys in chunks of 32. Per chunk: 4 WMMA for Q*K^T + 4 WMMA for P*V.
// or_mode: zero diagonal logits + multiply scaled logits by (1 - alpha^T).
// Output O merged-heads f16 [B, N, INNER].
// ---------------------------------------------------------------------------
__global__ void flash_attn_kernel(const _Float16* __restrict__ Qm, const _Float16* __restrict__ Km,
                                  const _Float16* __restrict__ Vt, const float* __restrict__ alpha,
                                  _Float16* __restrict__ O, int nseq, int or_mode) {
  __shared__ __attribute__((aligned(32))) _Float16 plds[4][16 * 32];
  int warp = threadIdx.x >> 5;
  int lane = threadIdx.x & 31;
  int nl = lane & 15, g = lane >> 4;
  int bh = blockIdx.y;
  int b = bh >> 3, h = bh & 7;
  int i0 = blockIdx.x * 64 + warp * 16;

  const _Float16* Qh = Qm + (size_t)bh * nseq * DHEAD;
  const _Float16* Kh = Km + (size_t)bh * nseq * DHEAD;
  const _Float16* Vh = Vt + (size_t)bh * DHEAD * nseq;
  const float* al = alpha + (size_t)b * nseq * nseq;
  _Float16* pl = plds[warp];

  // Q tile A-fragments (16 queries x 64 dims = two 16x32 fragments), held all loop
  v16h qa0 = load_afrag(Qh, DHEAD, i0, 0);
  v16h qa1 = load_afrag(Qh, DHEAD, i0, 32);

  v8f acc[4] = {};
  float mrow[8], lrow[8];
#pragma unroll
  for (int r = 0; r < 8; ++r) { mrow[r] = -1e30f; lrow[r] = 0.f; }

  for (int j0 = 0; j0 < nseq; j0 += 32) {
    // ---- scores: S = Q * K^T (K natural [keys][64] is exactly B-operand form)
    v8f s[2];
    {
      v8f z0 = {}, z1 = {};
      v16h kb;
      kb = load_bfrag(Kh, DHEAD, j0, 0);       s[0] = wmma_f16(qa0, kb, z0);
      kb = load_bfrag(Kh, DHEAD, j0, 32);      s[0] = wmma_f16(qa1, kb, s[0]);
      kb = load_bfrag(Kh, DHEAD, j0 + 16, 0);  s[1] = wmma_f16(qa0, kb, z1);
      kb = load_bfrag(Kh, DHEAD, j0 + 16, 32); s[1] = wmma_f16(qa1, kb, s[1]);
    }
    // ---- scale (+ OR-pass diagonal zero and (1 - alpha^T) rescale)
#pragma unroll
    for (int sub = 0; sub < 2; ++sub) {
      int kj = j0 + sub * 16 + nl;
#pragma unroll
      for (int r = 0; r < 8; ++r) {
        float sv = s[sub][r] * SCALEF;
        if (or_mode) {
          int qi = i0 + r + 8 * g;
          if (qi == kj) sv = 0.f;
          sv *= (1.f - al[(size_t)kj * nseq + qi]);
        }
        s[sub][r] = sv;
      }
    }
    // ---- online softmax (row = across 16 lanes of a half-wave)
#pragma unroll
    for (int r = 0; r < 8; ++r) {
      float mx = fmaxf(s[0][r], s[1][r]);
#pragma unroll
      for (int off = 8; off; off >>= 1) mx = fmaxf(mx, __shfl_xor(mx, off));
      float mn = fmaxf(mrow[r], mx);
      float corr = __expf(mrow[r] - mn);
      mrow[r] = mn;
      float p0 = __expf(s[0][r] - mn);
      float p1 = __expf(s[1][r] - mn);
      s[0][r] = p0;
      s[1][r] = p1;
      float rs = p0 + p1;
#pragma unroll
      for (int off = 8; off; off >>= 1) rs += __shfl_xor(rs, off);
      lrow[r] = lrow[r] * corr + rs;
#pragma unroll
      for (int t = 0; t < 4; ++t) acc[t][r] *= corr;
    }
    // ---- C-layout -> A-layout via per-wave LDS staging tile (16x32 f16)
#pragma unroll
    for (int sub = 0; sub < 2; ++sub)
#pragma unroll
      for (int r = 0; r < 8; ++r)
        pl[(r + 8 * g) * 32 + sub * 16 + nl] = (_Float16)s[sub][r];
    __syncthreads();
    v16h pa = load_afrag(pl, 32, 0, 0);
    __syncthreads();
    // ---- O += P * V  (V^T [64][N]: d-cols are B columns, keys are K dim)
#pragma unroll
    for (int t = 0; t < 4; ++t) {
      v16h vb = load_bfrag(Vh, nseq, 16 * t, j0);
      acc[t] = wmma_f16(pa, vb, acc[t]);
    }
  }

  // ---- normalize + store merged-heads f16
#pragma unroll
  for (int t = 0; t < 4; ++t) {
    int d = 16 * t + nl;
#pragma unroll
    for (int r = 0; r < 8; ++r) {
      int qi = i0 + r + 8 * g;
      O[((size_t)b * nseq + qi) * INNERC + h * DHEAD + d] =
          (_Float16)(acc[t][r] / lrow[r]);
    }
  }
}

// ---------------------------------------------------------------------------
extern "C" void kernel_launch(void* const* d_in, const int* in_sizes, int n_in,
                              void* d_out, int out_size, void* d_ws, size_t ws_size,
                              hipStream_t stream) {
  const float* x     = (const float*)d_in[0];
  const float* orq   = (const float*)d_in[1];
  const float* alpha = (const float*)d_in[2];
  const float* lnw   = (const float*)d_in[3];
  const float* lnb   = (const float*)d_in[4];
  const float* wqkv  = (const float*)d_in[5];
  const float* worq  = (const float*)d_in[6];
  const float* wout  = (const float*)d_in[7];
  float* out = (float*)d_out;

  const int Bsz = 8, Nseq = 1024;
  const int BN = Bsz * Nseq;  // 8192

  // workspace carve-up (f16), total ~70 MB
  _Float16* ws = (_Float16*)d_ws;
  _Float16* xn    = ws; ws += (size_t)BN * DIMC;
  _Float16* orqn  = ws; ws += (size_t)BN * DIMC;
  _Float16* wqkvT = ws; ws += (size_t)3 * INNERC * DIMC;
  _Float16* worqT = ws; ws += (size_t)INNERC * DIMC;
  _Float16* woutT = ws; ws += (size_t)DIMC * INNERC;
  _Float16* Qb    = ws; ws += (size_t)BN * INNERC;
  _Float16* Kb    = ws; ws += (size_t)BN * INNERC;
  _Float16* Vt    = ws; ws += (size_t)BN * INNERC;
  _Float16* ORQ   = ws; ws += (size_t)BN * INNERC;
  _Float16* Ostd  = ws; ws += (size_t)BN * INNERC;
  _Float16* Oor   = ws; ws += (size_t)BN * INNERC;

  // 1) LayerNorm + cast (x and ORquery together): one wave per row
  {
    int waves = 2 * BN;
    int threads = 256;  // 8 waves per block
    int blocks = (waves * 32 + threads - 1) / threads;
    ln_f16_kernel<<<blocks, threads, 0, stream>>>(x, orq, lnw, lnb, xn, orqn, BN);
  }
  // 2) weight transposes -> f16 [Ncol][K]
  {
    size_t n1 = (size_t)DIMC * 3 * INNERC;
    wtrans_kernel<<<(unsigned)((n1 + 255) / 256), 256, 0, stream>>>(wqkv, wqkvT, DIMC, 3 * INNERC);
    size_t n2 = (size_t)DIMC * INNERC;
    wtrans_kernel<<<(unsigned)((n2 + 255) / 256), 256, 0, stream>>>(worq, worqT, DIMC, INNERC);
    wtrans_kernel<<<(unsigned)((n2 + 255) / 256), 256, 0, stream>>>(wout, woutT, INNERC, DIMC);
  }
  // 3) QKV GEMM -> Q natural, K natural, V transposed (head-split)
  wmma_gemm_kernel<<<dim3(3 * INNERC / 64, BN / 128), 128, 0, stream>>>(
      xn, wqkvT, BN, 3 * INNERC, DIMC, Nseq, 0, Qb, Kb, Vt, nullptr);
  // 4) ORQ GEMM -> head-split natural
  wmma_gemm_kernel<<<dim3(INNERC / 64, BN / 128), 128, 0, stream>>>(
      orqn, worqT, BN, INNERC, DIMC, Nseq, 1, ORQ, nullptr, nullptr, nullptr);
  // 5) standard attention pass
  flash_attn_kernel<<<dim3(Nseq / 64, Bsz * HEADS), 128, 0, stream>>>(
      Qb, Kb, Vt, alpha, Ostd, Nseq, 0);
  // 6) OR attention pass (diag zero + (1 - alpha^T) rescale)
  flash_attn_kernel<<<dim3(Nseq / 64, Bsz * HEADS), 128, 0, stream>>>(
      ORQ, Kb, Vt, alpha, Oor, Nseq, 1);
  // 7/8) output projections -> f32 straight into d_out (concatenated)
  wmma_gemm_kernel<<<dim3(DIMC / 64, BN / 128), 128, 0, stream>>>(
      Ostd, woutT, BN, DIMC, INNERC, Nseq, 2, nullptr, nullptr, nullptr, out);
  wmma_gemm_kernel<<<dim3(DIMC / 64, BN / 128), 128, 0, stream>>>(
      Oor, woutT, BN, DIMC, INNERC, Nseq, 2, nullptr, nullptr, nullptr,
      out + (size_t)BN * DIMC);
}
